// Net_20461224198275
// MI455X (gfx1250) — matrix operand order repaired and checked by reading, not compiled, not run on hardware
//
#include <hip/hip_runtime.h>
#include <hip/hip_bf16.h>
#include <math.h>

// ---------------------------------------------------------------------------
// PointTransformerConv-style network, fused for MI455X (gfx1250, wave32).
// Dense layers use V_WMMA_F32_16X16X32_F16 (f16 in, f32 accumulate).
// All WMMA fragments are loaded as pairs of contiguous 128-bit loads
// (A from row-major f16 tiles, B from k-contiguous "fragment-ready" tiles).
// ---------------------------------------------------------------------------

#define NN    8192      // nodes
#define KK    32        // neighbors per node (contiguous dst runs)
#define HH    128
#define NECKN 512
#define GG    1024
#define FIN   59

typedef __attribute__((ext_vector_type(16))) _Float16 v16h;
typedef __attribute__((ext_vector_type(8)))  _Float16 v8h;
typedef __attribute__((ext_vector_type(8)))  float    v8f;

__device__ __forceinline__ v16h cat16(v8h lo, v8h hi) {
  return __builtin_shufflevector(lo, hi, 0, 1, 2, 3, 4, 5, 6, 7,
                                         8, 9, 10, 11, 12, 13, 14, 15);
}

// A fragment (16x32 f16, ISA 7.12.2): lane L holds row M=L%15? -> row L&15.
// halves a[0..7]  = K k0 + kh*8 + {0..7}
// halves a[8..15] = K k0 + 16 + kh*8 + {0..7}   (kh = lane>>4)
// Both runs contiguous in a row-major tile -> two 16B loads.
__device__ __forceinline__ v16h load_a_frag(const _Float16* base, int ldm,
                                            int row0, int k0, int lane) {
  const _Float16* p = base + (row0 + (lane & 15)) * ldm + k0 + (lane >> 4) * 8;
  v8h lo = *(const v8h*)p;
  v8h hi = *(const v8h*)(p + 16);
  return cat16(lo, hi);
}

// B fragment (32x16 f16): lane holds col = col0 + lane&15,
// halves b[0..7] = K k0 + kh*8 + {0..7}, b[8..15] = K k0 + 16 + kh*8 + {0..7} -- contiguous runs when the weight tile
// is staged transposed with k innermost: baseT[col][ldk].
__device__ __forceinline__ v16h load_b_fragT(const _Float16* baseT, int ldk,
                                             int k0, int col0, int lane) {
  const _Float16* p = baseT + (col0 + (lane & 15)) * ldk + k0 + (lane >> 4) * 8;
  v8h lo = *(const v8h*)p;
  v8h hi = *(const v8h*)(p + 16);
  return cat16(lo, hi);
}

// ---- generic fused GEMM: C = act(A[M,Kin] @ W[Kin,Nout] + bias) -----------
// block = 256 threads (8 waves), tile = 32 rows x 128 cols, K chunked by 64.
// mode: 0 = plain, 1 = +bias, 2 = +bias then BN(scale,shift)+ReLU
__global__ __launch_bounds__(256) void wmma_gemm_kernel(
    const float* __restrict__ A, const float* __restrict__ W,
    const float* __restrict__ bias, const float* __restrict__ gam,
    const float* __restrict__ bet, float* __restrict__ C,
    int M, int Kin, int Nout, int mode) {
  // padded to 72 halves (144B row stride): conflict-free half-wave access,
  // 16B alignment for ds_load_b128 preserved.
  __shared__ __attribute__((aligned(16))) _Float16 sA[32][72];
  __shared__ __attribute__((aligned(16))) _Float16 sWt[128][72]; // [col][k]
  const int rowBase = blockIdx.x * 32;
  const int colBase = blockIdx.y * 128;
  const int t = threadIdx.x, wave = t >> 5, lane = t & 31;
  v8f acc0 = {}; v8f acc1 = {};
  const int nch = (Kin + 63) >> 6;
  for (int kc = 0; kc < nch; ++kc) {
    // stage A chunk [32][64] (coalesced along k)
    for (int p = t; p < 32 * 64; p += 256) {
      int r = p >> 6, k = p & 63, gk = kc * 64 + k;
      float val = (gk < Kin) ? A[(size_t)(rowBase + r) * Kin + gk] : 0.f;
      sA[r][k] = (_Float16)val;
    }
    // stage W chunk transposed -> sWt[col][k] (global reads coalesced along col)
    for (int p = t; p < 64 * 128; p += 256) {
      int c = p & 127, k = p >> 7, gk = kc * 64 + k;
      float val = (gk < Kin) ? W[(size_t)gk * Nout + colBase + c] : 0.f;
      sWt[c][k] = (_Float16)val;
    }
    __syncthreads();
#pragma unroll
    for (int ks = 0; ks < 2; ++ks) {
      v16h bf = load_b_fragT(&sWt[0][0], 72, ks * 32, wave * 16, lane);
      v16h a0 = load_a_frag(&sA[0][0], 72, 0,  ks * 32, lane);
      v16h a1 = load_a_frag(&sA[0][0], 72, 16, ks * 32, lane);
      acc0 = __builtin_amdgcn_wmma_f32_16x16x32_f16(false, a0, false, bf,
                                                    (short)0, acc0, false, false);
      acc1 = __builtin_amdgcn_wmma_f32_16x16x32_f16(false, a1, false, bf,
                                                    (short)0, acc1, false, false);
    }
    __syncthreads();
  }
  // epilogue: C/D layout -> lane holds col=lane%16, rows 8*(lane/16)+r
  const int col = colBase + wave * 16 + (lane & 15);
  const int rb  = rowBase + 8 * (lane >> 4);
  float b = (mode >= 1 && bias) ? bias[col] : 0.f;
  float sc = 1.f, sh = 0.f;
  if (mode == 2) { sc = gam[col] * rsqrtf(1.f + 1e-5f); sh = bet[col]; }
#pragma unroll
  for (int r = 0; r < 8; ++r) {
    float t0 = acc0[r] + b;
    float t1 = acc1[r] + b;
    if (mode == 2) {
      t0 = fmaxf(t0 * sc + sh, 0.f);
      t1 = fmaxf(t1 * sc + sh, 0.f);
    }
    C[(size_t)(rb + r) * Nout + col]      = t0;
    C[(size_t)(rb + 16 + r) * Nout + col] = t1;
  }
}

// ---- fused per-node edge attention ----------------------------------------
// One block (256 thr, 8 waves) per destination node i: builds M[32,128] =
// xd[i]-xs[src]+delta in LDS (f16), alpha = bn_relu(M @ attnn_w) via WMMA
// (B frags read as global_load_b128 from a pre-transposed f16 weight copy),
// per-column softmax over the 32 edges, aggregate attn*(v[src]+delta).
#define MPAD 136   // 272B row stride: conflict-free, 16B aligned
#define FPAD 132
__global__ __launch_bounds__(256) void edge_attn_kernel(
    const float* __restrict__ xs, const float* __restrict__ xd,
    const float* __restrict__ vbuf, const float* __restrict__ pos,
    const float* __restrict__ nrm, const int* __restrict__ srcIdx,
    const float* __restrict__ posW, const float* __restrict__ posB,
    const float* __restrict__ posG, const float* __restrict__ posBB,
    const _Float16* __restrict__ attWT, const float* __restrict__ attB,
    const float* __restrict__ attG, const float* __restrict__ attBB,
    float* __restrict__ hOut) {
  __shared__ __attribute__((aligned(16))) _Float16 sM[KK][MPAD]; // WMMA A
  __shared__ float sVpd[KK][FPAD];   // v[src] + delta
  __shared__ float sAlpha[KK][FPAD];
  const int i = blockIdx.x;
  const int t = threadIdx.x;
  { // phase 1: gather + pos_nn MLP (K=6, pure VALU), 8 threads per edge
    int e  = t >> 3;
    int c0 = (t & 7) * 16;
    int s  = srcIdx[i * KK + e];
    float rel[6];
#pragma unroll
    for (int j = 0; j < 3; ++j) {
      rel[j]     = pos[i * 3 + j] - pos[s * 3 + j];
      rel[3 + j] = nrm[i * 3 + j] - nrm[s * 3 + j];
    }
#pragma unroll
    for (int cc = 0; cc < 16; ++cc) {
      int c = c0 + cc;
      float d = posB[c];
#pragma unroll
      for (int j = 0; j < 6; ++j) d += rel[j] * posW[j * HH + c];
      d = fmaxf(d * (posG[c] * rsqrtf(1.f + 1e-5f)) + posBB[c], 0.f);
      float m = xd[(size_t)i * HH + c] - xs[(size_t)s * HH + c] + d;
      sM[e][c]   = (_Float16)m;
      sVpd[e][c] = vbuf[(size_t)s * HH + c] + d;
    }
  }
  __syncthreads();
  // phase 2: alpha = M[32,128] @ attW[128,128] — 8 WMMAs per wave
  const int wave = t >> 5, lane = t & 31;
  v8f acc0 = {}; v8f acc1 = {};
#pragma unroll
  for (int ks = 0; ks < 4; ++ks) {
    v16h bf = load_b_fragT(attWT, HH, ks * 32, wave * 16, lane);
    v16h a0 = load_a_frag(&sM[0][0], MPAD, 0,  ks * 32, lane);
    v16h a1 = load_a_frag(&sM[0][0], MPAD, 16, ks * 32, lane);
    acc0 = __builtin_amdgcn_wmma_f32_16x16x32_f16(false, a0, false, bf,
                                                  (short)0, acc0, false, false);
    acc1 = __builtin_amdgcn_wmma_f32_16x16x32_f16(false, a1, false, bf,
                                                  (short)0, acc1, false, false);
  }
  { // bn_relu epilogue into LDS
    int col = wave * 16 + (lane & 15);
    int rb  = 8 * (lane >> 4);
    float b  = attB[col];
    float sc = attG[col] * rsqrtf(1.f + 1e-5f);
    float sh = attBB[col];
#pragma unroll
    for (int r = 0; r < 8; ++r) {
      sAlpha[rb + r][col]      = fmaxf((acc0[r] + b) * sc + sh, 0.f);
      sAlpha[rb + 16 + r][col] = fmaxf((acc1[r] + b) * sc + sh, 0.f);
    }
  }
  __syncthreads();
  // phase 3: per-(node,feature) softmax over 32 neighbors + weighted sum
  if (t < HH) {
    float mx = -3.4e38f;
#pragma unroll
    for (int e = 0; e < KK; ++e) mx = fmaxf(mx, sAlpha[e][t]);
    float ex[KK];
    float den = 0.f;
#pragma unroll
    for (int e = 0; e < KK; ++e) { ex[e] = __expf(sAlpha[e][t] - mx); den += ex[e]; }
    float inv = 1.f / (den + 1e-16f);
    float acc = 0.f;
#pragma unroll
    for (int e = 0; e < KK; ++e) acc += ex[e] * sVpd[e][t];
    hOut[(size_t)i * HH + t] = acc * inv;
  }
}

// ---- pool_batch relabel (generic cumsum-of-changes) -> group starts -------
__global__ __launch_bounds__(1024) void relabel_kernel(const int* __restrict__ pb,
                                                       int* __restrict__ gstart) {
  __shared__ int cnt[1024];
  const int t = threadIdx.x;
  const int base = t * (NN / 1024);
  int local = 0;
  for (int j = 0; j < NN / 1024; ++j) {
    int i = base + j;
    if (i > 0 && pb[i] != pb[i - 1]) local++;
  }
  cnt[t] = local;
  __syncthreads();
  if (t == 0) {
    int run = 0;
    for (int k = 0; k < 1024; ++k) { int c = cnt[k]; cnt[k] = run; run += c; }
  }
  __syncthreads();
  int lbl = cnt[t];
  for (int j = 0; j < NN / 1024; ++j) {
    int i = base + j;
    if (i == 0) gstart[0] = 0;
    else if (pb[i] != pb[i - 1]) { lbl++; gstart[lbl] = i; }
  }
  if (t == 1023) gstart[lbl + 1] = NN;
}

// ---- group max-pool over hn + pooled mask ---------------------------------
__global__ __launch_bounds__(256) void pool_kernel(const float* __restrict__ hn,
    const float* __restrict__ mask_t, const int* __restrict__ gstart,
    float* __restrict__ gfeat, float* __restrict__ mask_g) {
  const int g = blockIdx.x;
  const int s = gstart[g], e = gstart[g + 1];
  for (int c = threadIdx.x; c < NECKN; c += 256) {
    float mx = -3.4e38f;
    for (int r = s; r < e; ++r) mx = fmaxf(mx, hn[(size_t)r * NECKN + c]);
    gfeat[(size_t)g * NECKN + c] = mx;
  }
  if (threadIdx.x == 0) {
    float mm = -3.4e38f;
    for (int r = s; r < e; ++r) mm = fmaxf(mm, mask_t[r]);
    mask_g[g] = mm;
  }
}

// ---- head: out[g] = mask * (gm1[g] . mlp2_w + b) --------------------------
__global__ __launch_bounds__(256) void head_kernel(const float* __restrict__ gm1,
    const float* __restrict__ w2, const float* __restrict__ b2,
    const float* __restrict__ mask_g, float* __restrict__ out) {
  __shared__ float red[256];
  const int g = blockIdx.x, t = threadIdx.x;
  red[t] = gm1[(size_t)g * 256 + t] * w2[t];
  __syncthreads();
  for (int s = 128; s > 0; s >>= 1) {
    if (t < s) red[t] += red[t + s];
    __syncthreads();
  }
  if (t == 0) out[g] = (mask_g[g] == 1.0f) ? (red[0] + b2[0]) : 0.f;
}

// ---- transpose f32 [K,N] -> fragment-ready f16 [N][K] ---------------------
__global__ void cvt_transpose_f16_kernel(const float* __restrict__ in,
                                         _Float16* __restrict__ outT,
                                         int K, int N) {
  int i = blockIdx.x * blockDim.x + threadIdx.x;
  if (i < K * N) {
    int k = i / N, c = i % N;               // coalesced read of in[k][c]
    outT[(size_t)c * K + k] = (_Float16)in[i];
  }
}

extern "C" void kernel_launch(void* const* d_in, const int* in_sizes, int n_in,
                              void* d_out, int out_size, void* d_ws, size_t ws_size,
                              hipStream_t stream) {
  const float* x          = (const float*)d_in[0];
  const float* pos        = (const float*)d_in[1];
  const float* nrm        = (const float*)d_in[2];
  const float* mask_t     = (const float*)d_in[3];
  const int*   pool_batch = (const int*)d_in[4];
  const int*   src        = (const int*)d_in[5];
  // d_in[6] = dst: by construction dst[e] == e / K (contiguous runs) — fused.
  const float* lin_w   = (const float*)d_in[7];
  const float* lin_b   = (const float*)d_in[8];
  const float* src_w   = (const float*)d_in[9];
  const float* dst_w   = (const float*)d_in[10];
  const float* posnn_w = (const float*)d_in[11];
  const float* posnn_b = (const float*)d_in[12];
  const float* posnn_g = (const float*)d_in[13];
  const float* posnn_bb= (const float*)d_in[14];
  const float* attnn_w = (const float*)d_in[15];
  const float* attnn_b = (const float*)d_in[16];
  const float* attnn_g = (const float*)d_in[17];
  const float* attnn_bb= (const float*)d_in[18];
  const float* neck_w  = (const float*)d_in[19];
  const float* neck_b  = (const float*)d_in[20];
  const float* neck_g  = (const float*)d_in[21];
  const float* neck_bb = (const float*)d_in[22];
  const float* mlp1_w  = (const float*)d_in[23];
  const float* mlp1_b  = (const float*)d_in[24];
  const float* mlp1_g  = (const float*)d_in[25];
  const float* mlp1_bb = (const float*)d_in[26];
  const float* mlp2_w  = (const float*)d_in[27];
  const float* mlp2_b  = (const float*)d_in[28];
  float* out = (float*)d_out;

  // Workspace layout (lifetimes overlap where buffers are dead):
  //   [0, 16MB)   : v,xs,xd (phase 1)  then hn [N,512] (phase 2)
  //   [16, 20MB)  : h [N,128]
  //   [20, 22MB)  : gfeat [G,512]
  //   [22, 23MB)  : gm1 [G,256]
  //   [23MB, ...) : attWT (f16, transposed), gstart, mask_g
  char* ws = (char*)d_ws;
  float*    vbuf   = (float*)(ws + 0);
  float*    xsb    = (float*)(ws + ((size_t)4  << 20));
  float*    xdb    = (float*)(ws + ((size_t)8  << 20));
  float*    hn     = (float*)(ws + 0);
  float*    h      = (float*)(ws + ((size_t)16 << 20));
  float*    gfeat  = (float*)(ws + ((size_t)20 << 20));
  float*    gm1    = (float*)(ws + ((size_t)22 << 20));
  _Float16* attWT  = (_Float16*)(ws + ((size_t)23 << 20));
  int*      gstart = (int*)(ws + ((size_t)23 << 20) + 65536);
  float*    maskg  = (float*)(ws + ((size_t)23 << 20) + 131072);

  // group boundaries from pool_batch (generic relabel)
  relabel_kernel<<<1, 1024, 0, stream>>>(pool_batch, gstart);
  // fragment-ready f16 transpose of attnn_w (k contiguous per output column)
  cvt_transpose_f16_kernel<<<(HH * HH + 255) / 256, 256, 0, stream>>>(
      attnn_w, attWT, HH, HH);

  // projections: v = x@lin_w + lin_b ; xs = x@src_w ; xd = x@dst_w
  wmma_gemm_kernel<<<dim3(NN / 32, 1), 256, 0, stream>>>(
      x, lin_w, lin_b, nullptr, nullptr, vbuf, NN, FIN, HH, 1);
  wmma_gemm_kernel<<<dim3(NN / 32, 1), 256, 0, stream>>>(
      x, src_w, nullptr, nullptr, nullptr, xsb, NN, FIN, HH, 0);
  wmma_gemm_kernel<<<dim3(NN / 32, 1), 256, 0, stream>>>(
      x, dst_w, nullptr, nullptr, nullptr, xdb, NN, FIN, HH, 0);

  // fused edge gather -> pos_nn -> attn GEMM -> segment softmax -> aggregate
  edge_attn_kernel<<<NN, 256, 0, stream>>>(
      xsb, xdb, vbuf, pos, nrm, src,
      posnn_w, posnn_b, posnn_g, posnn_bb,
      attWT, attnn_b, attnn_g, attnn_bb, h);

  // neck: hn = bn_relu(h @ neck_w + neck_b)   [8192,512]
  wmma_gemm_kernel<<<dim3(NN / 32, NECKN / 128), 256, 0, stream>>>(
      h, neck_w, neck_b, neck_g, neck_bb, hn, NN, HH, NECKN, 2);

  // residue max-pool + pooled mask
  pool_kernel<<<GG, 256, 0, stream>>>(hn, mask_t, gstart, gfeat, maskg);

  // mlp1: gm1 = bn_relu(gfeat @ mlp1_w + mlp1_b)   [1024,256]
  wmma_gemm_kernel<<<dim3(GG / 32, 256 / 128), 256, 0, stream>>>(
      gfeat, mlp1_w, mlp1_b, mlp1_g, mlp1_bb, gm1, GG, NECKN, 256, 2);

  // head: out = mask * (gm1 @ mlp2_w + mlp2_b)
  head_kernel<<<GG, 256, 0, stream>>>(gm1, mlp2_w, mlp2_b, maskg, out);
}
